// Qwen2_5OmniAudioEncoderLayer_5626407157997
// MI455X (gfx1250) — compile-verified, matrix operand order checked by reading, not run on hardware
//
#include <hip/hip_runtime.h>
#include <cstddef>
#include <cstdint>

// ---------------------------------------------------------------------------
// Types for CDNA5 WMMA (wave32): 16x16x32 bf16 -> f32, plus TDM descriptors.
// ---------------------------------------------------------------------------
typedef __bf16 bf16_t;
typedef __attribute__((ext_vector_type(16))) __bf16    v16bf;
typedef __attribute__((ext_vector_type(8)))  float     v8f;
typedef __attribute__((ext_vector_type(4)))  uint32_t  v4u;
typedef __attribute__((ext_vector_type(8)))  int       v8i;
typedef __attribute__((ext_vector_type(4)))  int       v4i;

#if defined(__has_builtin)
#if __has_builtin(__builtin_amdgcn_tensor_load_to_lds) && __has_builtin(__builtin_amdgcn_s_wait_tensorcnt)
#define HAVE_TDM 1
#endif
#endif
#ifndef HAVE_TDM
#define HAVE_TDM 0
#endif

__device__ __forceinline__ v8f zero8() {
    v8f z;
#pragma unroll
    for (int i = 0; i < 8; ++i) z[i] = 0.0f;
    return z;
}

// A/B fragment K-index mapping for 16-bit 16x32 operands (ISA 7.12.2):
// lanes 0-15 hold K=[0..7,16..23], lanes 16-31 hold K=[8..15,24..31].
#define KK(e, hi) ((hi) * 8 + (e) + (((e) >> 3) << 3))

// ---------------------------------------------------------------------------
// TDM: 2D tile load Global(bf16) -> LDS.  D# per ISA cdna5 §8.3/8.4.
// tensor dims are set == tile dims (all tiles here are fully in-bounds).
// padcfg packs pad_enable/pad_interval/pad_amount bits of group1 word0.
// This toolchain's builtin is the 6-arg clang-23 form:
//   (uint32x4 g0, int32x8 g1, int32x4 g2, int32x4 g3, int32x8, i32 cpol)
// ---------------------------------------------------------------------------
#if HAVE_TDM
__device__ __forceinline__ void tdm_load_2d(
    uint32_t lds_byte, const bf16_t* gptr,
    uint32_t tile0, uint32_t tile1, uint32_t stride0_elems, uint32_t padcfg)
{
    const uint64_t ga = (uint64_t)(uintptr_t)gptr;
    v4u g0;
    g0[0] = 1u;                                        // count=1, user mode
    g0[1] = lds_byte;                                  // lds_addr [63:32]
    g0[2] = (uint32_t)(ga & 0xFFFFFFFFu);              // global_addr lo
    g0[3] = (uint32_t)((ga >> 32) & 0x01FFFFFFu)       // global_addr hi (57b)
          | (2u << 30);                                // type=2 ("image")
    v8i g1;
    g1[0] = (int)((1u << 16) | padcfg);                // data_size=1 (2B) + pad cfg
    g1[1] = (int)((tile0 & 0xFFFFu) << 16);            // tensor_dim0 lo16 @ [63:48]
    g1[2] = (int)(((tile0 >> 16) & 0xFFFFu)            // tensor_dim0 hi16
          | ((tile1 & 0xFFFFu) << 16));                // tensor_dim1 lo16
    g1[3] = (int)(((tile1 >> 16) & 0xFFFFu)            // tensor_dim1 hi16
          | ((tile0 & 0xFFFFu) << 16));                // tile_dim0
    g1[4] = (int)(tile1 & 0xFFFFu);                    // tile_dim1 (tile_dim2=0)
    g1[5] = (int)stride0_elems;                        // tensor_dim0_stride lo32
    g1[6] = 0;                                         // stride hi16, dim1_stride lo16
    g1[7] = 0;                                         // dim1_stride hi32
    const v4i z4 = {0, 0, 0, 0};
    const v8i z8 = {0, 0, 0, 0, 0, 0, 0, 0};
    __builtin_amdgcn_tensor_load_to_lds(g0, g1, z4, z4, z8, 0);
}
// pad configs: pad 1 dword after every 2^(c+1) dwords stored
#define PAD_A  ((1u << 20) | (3u << 22))   // after 16 dwords (row=32 bf16) -> pitch 34
#define PAD_B  ((1u << 20) | (5u << 22))   // after 64 dwords (row=128 bf16) -> pitch 130
#define PAD_KV ((1u << 20) | (4u << 22))   // after 32 dwords (row=64 bf16) -> pitch 66
#endif

__device__ __forceinline__ uint32_t lds_byte_addr(const void* p) {
    // Flat LDS aperture: addr[31:0] is the wave-relative LDS byte address.
    return (uint32_t)(uintptr_t)p;
}

// ---------------------------------------------------------------------------
// fp32 -> bf16 conversion (weights, once per call).
// ---------------------------------------------------------------------------
__global__ __launch_bounds__(256) void cvt_kernel(
    const float* __restrict__ in, bf16_t* __restrict__ out, int n)
{
    int i = blockIdx.x * 256 + threadIdx.x;
    if (i < n) out[i] = (bf16_t)in[i];
}

// ---------------------------------------------------------------------------
// LayerNorm: one block (256 threads) per row; fp32 in, bf16 out.
// ---------------------------------------------------------------------------
__global__ __launch_bounds__(256) void ln_kernel(
    const float* __restrict__ x, const float* __restrict__ g,
    const float* __restrict__ b, bf16_t* __restrict__ y, int Dm)
{
    __shared__ float red[256];
    const int row = blockIdx.x;
    const int tid = threadIdx.x;
    const float* xr = x + (size_t)row * Dm;

    float s = 0.f, s2 = 0.f;
    for (int i = tid; i < Dm; i += 256) { float v = xr[i]; s += v; s2 += v * v; }

    red[tid] = s; __syncthreads();
    for (int off = 128; off > 0; off >>= 1) {
        if (tid < off) red[tid] += red[tid + off];
        __syncthreads();
    }
    const float mean = red[0] / (float)Dm;
    __syncthreads();

    red[tid] = s2; __syncthreads();
    for (int off = 128; off > 0; off >>= 1) {
        if (tid < off) red[tid] += red[tid + off];
        __syncthreads();
    }
    const float var = red[0] / (float)Dm - mean * mean;
    const float inv = rsqrtf(var + 1e-5f);

    bf16_t* yr = y + (size_t)row * Dm;
    for (int i = tid; i < Dm; i += 256)
        yr[i] = (bf16_t)((xr[i] - mean) * inv * g[i] + b[i]);
}

// ---------------------------------------------------------------------------
// WMMA GEMM: C[M,N] = act( A[M,K](bf16) @ B[K,N](bf16) + bias (+ res) )
// Block tile 128x128, 8 waves as 4(M) x 2(N); each wave owns 32x64
// -> 8 WMMAs per 32-K step.  A/B tiles staged by the Tensor Data Mover
// (double-buffered, wave 0 issues, s_wait_tensorcnt + barrier sync).
// M%128==0, N%128==0, K%32==0 (holds for all shapes here).
// ---------------------------------------------------------------------------
template <bool OUTBF16, bool DOGELU, bool RES>
__global__ __launch_bounds__(256) void gemm_wmma(
    const bf16_t* __restrict__ A, const bf16_t* __restrict__ B,
    const float* __restrict__ bias, const float* __restrict__ res,
    void* __restrict__ Cout, int M, int N, int K)
{
    constexpr int BM = 128, BN = 128, KT = 32;
    constexpr int AP = 34, BP = 130;           // padded LDS pitches (bf16 elems)
    __shared__ bf16_t As[2][BM * AP];
    __shared__ bf16_t Bs[2][KT * BP];

    const int tid  = threadIdx.x;
    const int wave = tid >> 5;
    const int lane = tid & 31;
    const int hi   = (lane >= 16) ? 1 : 0;
    const int l15  = lane & 15;
    const int wm   = wave & 3;                 // 4 waves along M (32 rows each)
    const int wn   = wave >> 2;                // 2 waves along N (64 cols each)
    const int m0   = blockIdx.y * BM;
    const int n0   = blockIdx.x * BN;

    v8f acc[2][4];
#pragma unroll
    for (int r2 = 0; r2 < 2; ++r2)
#pragma unroll
        for (int t = 0; t < 4; ++t) acc[r2][t] = zero8();

    const int ntiles = K / KT;

#if HAVE_TDM
    const uint32_t asb[2] = { lds_byte_addr(&As[0][0]), lds_byte_addr(&As[1][0]) };
    const uint32_t bsb[2] = { lds_byte_addr(&Bs[0][0]), lds_byte_addr(&Bs[1][0]) };
    if (wave == 0) {
        tdm_load_2d(asb[0], A + (size_t)m0 * K, KT, BM, (uint32_t)K, PAD_A);
        tdm_load_2d(bsb[0], B + (size_t)0 * N + n0, BN, KT, (uint32_t)N, PAD_B);
    }
#endif

    for (int i = 0; i < ntiles; ++i) {
        const int cur = i & 1;
#if HAVE_TDM
        if (wave == 0) {
            if (i + 1 < ntiles) {
                const int k1 = (i + 1) * KT;
                tdm_load_2d(asb[cur ^ 1], A + (size_t)m0 * K + k1, KT, BM, (uint32_t)K, PAD_A);
                tdm_load_2d(bsb[cur ^ 1], B + (size_t)k1 * N + n0, BN, KT, (uint32_t)N, PAD_B);
                __builtin_amdgcn_s_wait_tensorcnt((short)2);   // tile i landed
            } else {
                __builtin_amdgcn_s_wait_tensorcnt((short)0);
            }
        }
        __syncthreads();
#else
        // Manual staging fallback.
        {
            const int k0 = i * KT;
#pragma unroll
            for (int j = 0; j < 16; ++j) {     // A: 128x32 = 4096 elems
                int idx = j * 256 + tid;
                int r = idx >> 5, c = idx & 31;
                As[cur][r * AP + c] = A[(size_t)(m0 + r) * K + (k0 + c)];
            }
#pragma unroll
            for (int j = 0; j < 16; ++j) {     // B: 32x128 = 4096 elems
                int idx = j * 256 + tid;
                int r = idx >> 7, c = idx & 127;
                Bs[cur][r * BP + c] = B[(size_t)(k0 + r) * N + (n0 + c)];
            }
        }
        __syncthreads();
#endif
        const bf16_t* as = As[cur];
        const bf16_t* bs = Bs[cur];

        v16bf a[2];
#pragma unroll
        for (int r2 = 0; r2 < 2; ++r2) {
            const int arow = wm * 32 + r2 * 16 + l15;
#pragma unroll
            for (int e = 0; e < 16; ++e) a[r2][e] = as[arow * AP + KK(e, hi)];
        }
#pragma unroll
        for (int t = 0; t < 4; ++t) {
            v16bf bfr;
#pragma unroll
            for (int e = 0; e < 16; ++e)
                bfr[e] = bs[KK(e, hi) * BP + wn * 64 + t * 16 + l15];
#pragma unroll
            for (int r2 = 0; r2 < 2; ++r2)
                acc[r2][t] = __builtin_amdgcn_wmma_f32_16x16x32_bf16(
                    false, a[r2], false, bfr, (short)0, acc[r2][t], false, false);
        }
        __syncthreads();
    }

    // Epilogue: bias (+res) (+gelu), store fp32 or bf16.
#pragma unroll
    for (int r2 = 0; r2 < 2; ++r2) {
#pragma unroll
        for (int t = 0; t < 4; ++t) {
#pragma unroll
            for (int r = 0; r < 8; ++r) {
                const int m = m0 + wm * 32 + r2 * 16 + r + hi * 8;
                const int n = n0 + wn * 64 + t * 16 + l15;
                float c = acc[r2][t][r];
                if (bias) c += bias[n];
                if constexpr (RES) c += res[(size_t)m * N + n];
                if constexpr (DOGELU) c = 0.5f * c * (1.0f + erff(c * 0.70710678118654752f));
                if constexpr (OUTBF16)
                    ((bf16_t*)Cout)[(size_t)m * N + n] = (bf16_t)c;
                else
                    ((float*)Cout)[(size_t)m * N + n] = c;
            }
        }
    }
}

// ---------------------------------------------------------------------------
// Block-diagonal flash attention with WMMA, bf16 Q/K/V in, bf16 out.
// grid = (H, nseg, seglen/128); 8 waves; each wave owns a 16-row Q tile.
// K/V 32x64 tiles staged by TDM; per 32-key iteration:
//   S(16x32) = Q @ K^T (4 WMMAs), online softmax (shfl_xor butterflies),
//   O += P @ V (4 WMMAs, P relayout C->A via wave-local LDS).
// ---------------------------------------------------------------------------
__global__ __launch_bounds__(256) void attn_kernel(
    const bf16_t* __restrict__ Q, const bf16_t* __restrict__ Km,
    const bf16_t* __restrict__ V, const int* __restrict__ cu,
    bf16_t* __restrict__ O)
{
    constexpr int Dm = 1280, HDm = 64;
    constexpr int KP = 66;                     // padded K/V LDS pitch
    __shared__ bf16_t Ks[32 * KP];
    __shared__ bf16_t Vs[32 * KP];
    __shared__ bf16_t Pbuf[8][16][32 + 2];

    const int h    = blockIdx.x;
    const int seg  = blockIdx.y;
    const int tid  = threadIdx.x;
    const int wave = tid >> 5;
    const int lane = tid & 31;
    const int hi   = (lane >= 16) ? 1 : 0;
    const int l15  = lane & 15;

    const int kstart = cu[seg];
    const int kend   = cu[seg + 1];
    const int q0     = kstart + blockIdx.z * 128 + wave * 16;

    // Q fragments for both K-chunks of head_dim (0..31, 32..63).
    v16bf qf[2];
#pragma unroll
    for (int c = 0; c < 2; ++c)
#pragma unroll
        for (int e = 0; e < 16; ++e)
            qf[c][e] = Q[(size_t)(q0 + l15) * Dm + h * HDm + c * 32 + KK(e, hi)];

    v8f ao[4];
#pragma unroll
    for (int t = 0; t < 4; ++t) ao[t] = zero8();
    float mi[8], li[8];
#pragma unroll
    for (int r = 0; r < 8; ++r) { mi[r] = -3.0e38f; li[r] = 0.0f; }

#if HAVE_TDM
    const uint32_t ksb = lds_byte_addr(&Ks[0]);
    const uint32_t vsb = lds_byte_addr(&Vs[0]);
#endif

    for (int kb = kstart; kb < kend; kb += 32) {
#if HAVE_TDM
        if (wave == 0) {
            tdm_load_2d(ksb, Km + (size_t)kb * Dm + h * HDm, HDm, 32, (uint32_t)Dm, PAD_KV);
            tdm_load_2d(vsb, V  + (size_t)kb * Dm + h * HDm, HDm, 32, (uint32_t)Dm, PAD_KV);
            __builtin_amdgcn_s_wait_tensorcnt((short)0);
        }
        __syncthreads();
#else
#pragma unroll
        for (int i = 0; i < 8; ++i) {          // 32x64 = 2048 elems each
            int idx = i * 256 + tid;
            int r = idx >> 6, c = idx & 63;
            size_t go = (size_t)(kb + r) * Dm + h * HDm + c;
            Ks[r * KP + c] = Km[go];
            Vs[r * KP + c] = V[go];
        }
        __syncthreads();
#endif

        // S = Q @ K^T (two 16x16 n-tiles over 32 keys).
        v8f s[2];
#pragma unroll
        for (int t = 0; t < 2; ++t) {
            s[t] = zero8();
#pragma unroll
            for (int c = 0; c < 2; ++c) {
                v16bf kf;
#pragma unroll
                for (int e = 0; e < 16; ++e)
                    kf[e] = Ks[(t * 16 + l15) * KP + c * 32 + KK(e, hi)];
                s[t] = __builtin_amdgcn_wmma_f32_16x16x32_bf16(
                    false, qf[c], false, kf, (short)0, s[t], false, false);
            }
        }

        // Scale + key tail mask (inert for uniform 512-token segments).
#pragma unroll
        for (int t = 0; t < 2; ++t) {
            const bool oob = (kb + t * 16 + l15) >= kend;
#pragma unroll
            for (int r = 0; r < 8; ++r) {
                float v = s[t][r] * 0.125f;    // 1/sqrt(64)
                s[t][r] = oob ? -3.0e38f : v;
            }
        }

        // Online softmax per row (row stats live in the owning 16-lane half).
#pragma unroll
        for (int r = 0; r < 8; ++r) {
            float rm = fmaxf(s[0][r], s[1][r]);
#pragma unroll
            for (int off = 1; off < 16; off <<= 1)
                rm = fmaxf(rm, __shfl_xor(rm, off, 16));
            const float mnew  = fmaxf(mi[r], rm);
            const float alpha = __expf(mi[r] - mnew);
            const float p0 = __expf(s[0][r] - mnew);
            const float p1 = __expf(s[1][r] - mnew);
            float rs = p0 + p1;
#pragma unroll
            for (int off = 1; off < 16; off <<= 1)
                rs += __shfl_xor(rs, off, 16);
            li[r] = li[r] * alpha + rs;
            mi[r] = mnew;
#pragma unroll
            for (int t = 0; t < 4; ++t) ao[t][r] *= alpha;

            const int row = r + hi * 8;
            Pbuf[wave][row][l15]      = (bf16_t)p0;
            Pbuf[wave][row][16 + l15] = (bf16_t)p1;
        }
        asm volatile("s_wait_dscnt 0" ::: "memory");

        // P fragment (16x32 A operand) and O += P @ V.
        v16bf pf;
#pragma unroll
        for (int e = 0; e < 16; ++e) pf[e] = Pbuf[wave][l15][KK(e, hi)];
#pragma unroll
        for (int t = 0; t < 4; ++t) {
            v16bf vf;
#pragma unroll
            for (int e = 0; e < 16; ++e)
                vf[e] = Vs[KK(e, hi) * KP + t * 16 + l15];
            ao[t] = __builtin_amdgcn_wmma_f32_16x16x32_bf16(
                false, pf, false, vf, (short)0, ao[t], false, false);
        }
        __syncthreads();
    }

    // Normalize and write [S, H*HD] == reshape(S, D), bf16.
#pragma unroll
    for (int t = 0; t < 4; ++t)
#pragma unroll
        for (int r = 0; r < 8; ++r) {
            const int m = q0 + r + hi * 8;
            O[(size_t)m * Dm + h * HDm + t * 16 + l15] = (bf16_t)(ao[t][r] / li[r]);
        }
}

// ---------------------------------------------------------------------------
// Host launcher.
// ---------------------------------------------------------------------------
extern "C" void kernel_launch(void* const* d_in, const int* in_sizes, int n_in,
                              void* d_out, int out_size, void* d_ws, size_t ws_size,
                              hipStream_t stream) {
    const float* hidden = (const float*)d_in[0];
    const int*   cu     = (const int*)  d_in[1];
    const float* Wq = (const float*)d_in[2];  const float* bq = (const float*)d_in[3];
    const float* Wk = (const float*)d_in[4];
    const float* Wv = (const float*)d_in[5];  const float* bv = (const float*)d_in[6];
    const float* Wo = (const float*)d_in[7];  const float* bo = (const float*)d_in[8];
    const float* ln1g = (const float*)d_in[9];  const float* ln1b = (const float*)d_in[10];
    const float* Wf1 = (const float*)d_in[11]; const float* bf1 = (const float*)d_in[12];
    const float* Wf2 = (const float*)d_in[13]; const float* bf2 = (const float*)d_in[14];
    const float* ln2g = (const float*)d_in[15]; const float* ln2b = (const float*)d_in[16];

    const int Dm = 1280, FFNm = 5120, Hh = 20;
    const int S      = in_sizes[0] / Dm;     // 4096
    const int nseg   = in_sizes[1] - 1;      // 8
    const int seglen = S / nseg;             // 512

    const size_t DD = (size_t)Dm * Dm;
    const size_t DF = (size_t)Dm * FFNm;
    const size_t SD = (size_t)S * Dm;
    const size_t SF = (size_t)S * FFNm;

    bf16_t* wb    = (bf16_t*)d_ws;
    bf16_t* Wq_b  = wb;  wb += DD;
    bf16_t* Wk_b  = wb;  wb += DD;
    bf16_t* Wv_b  = wb;  wb += DD;
    bf16_t* Wo_b  = wb;  wb += DD;
    bf16_t* Wf1_b = wb;  wb += DF;
    bf16_t* Wf2_b = wb;  wb += DF;
    bf16_t* xbf   = wb;  wb += SD;           // LN1 out (reused for LN2 out)
    bf16_t* qbf   = wb;  wb += SD;
    bf16_t* kbf   = wb;  wb += SD;
    bf16_t* vbf   = wb;  wb += SD;
    bf16_t* attbf = wb;  wb += SD;
    bf16_t* gbf   = wb;  wb += SF;           // GELU(FFN1), bf16
    float*  hbuf  = (float*)wb;              // residual stream, fp32
    float*  out   = (float*)d_out;

    const dim3 blk(256);
    const dim3 gD(Dm / 128, S / 128);        // (10, 32)
    const dim3 gF(FFNm / 128, S / 128);      // (40, 32)
    const dim3 gA(Hh, nseg, seglen / 128);   // (20, 8, 4)

    // One-time fp32 -> bf16 weight conversion (deterministic, every call).
    cvt_kernel<<<dim3((int)((DD + 255) / 256)), blk, 0, stream>>>(Wq,  Wq_b,  (int)DD);
    cvt_kernel<<<dim3((int)((DD + 255) / 256)), blk, 0, stream>>>(Wk,  Wk_b,  (int)DD);
    cvt_kernel<<<dim3((int)((DD + 255) / 256)), blk, 0, stream>>>(Wv,  Wv_b,  (int)DD);
    cvt_kernel<<<dim3((int)((DD + 255) / 256)), blk, 0, stream>>>(Wo,  Wo_b,  (int)DD);
    cvt_kernel<<<dim3((int)((DF + 255) / 256)), blk, 0, stream>>>(Wf1, Wf1_b, (int)DF);
    cvt_kernel<<<dim3((int)((DF + 255) / 256)), blk, 0, stream>>>(Wf2, Wf2_b, (int)DF);

    // LN1
    ln_kernel<<<S, blk, 0, stream>>>(hidden, ln1g, ln1b, xbf, Dm);
    // Q/K/V projections (bf16 out)
    gemm_wmma<true,false,false><<<gD, blk, 0, stream>>>(xbf, Wq_b, bq, nullptr, qbf, S, Dm, Dm);
    gemm_wmma<true,false,false><<<gD, blk, 0, stream>>>(xbf, Wk_b, nullptr, nullptr, kbf, S, Dm, Dm);
    gemm_wmma<true,false,false><<<gD, blk, 0, stream>>>(xbf, Wv_b, bv, nullptr, vbf, S, Dm, Dm);
    // Block-diagonal flash attention (bf16 out)
    attn_kernel<<<gA, blk, 0, stream>>>(qbf, kbf, vbf, cu, attbf);
    // Output projection + first residual (fp32 h = hidden + att@Wo + bo)
    gemm_wmma<false,false,true><<<gD, blk, 0, stream>>>(attbf, Wo_b, bo, hidden, hbuf, S, Dm, Dm);
    // LN2 (bf16 out)
    ln_kernel<<<S, blk, 0, stream>>>(hbuf, ln2g, ln2b, xbf, Dm);
    // FFN1 + exact GELU (bf16 out)
    gemm_wmma<true,true,false><<<gF, blk, 0, stream>>>(xbf, Wf1_b, bf1, nullptr, gbf, S, FFNm, Dm);
    // FFN2 + second residual -> final fp32 output
    gemm_wmma<false,false,true><<<gD, blk, 0, stream>>>(gbf, Wf2_b, bf2, hbuf, out, S, Dm, FFNm);
}
